// CustomLSTM_33990371181570
// MI455X (gfx1250) — compile-verified, matrix operand order checked by reading, not run on hardware
//
#include <hip/hip_runtime.h>
#include <stdint.h>

// ---------------------------------------------------------------------------
// CDNA5 (gfx1250) fused LSTM
//   - per-timestep GEMM on v_wmma_f32_16x16x32_bf16
//   - B operands staged to LDS via Tensor Data Mover (tensor_load_to_lds,
//     TENSORcnt), double buffered, wave0-issued, overlapped with compute
//   - all 8 B fragments of a k-tile loaded into distinct VGPRs before the
//     WMMA batch so ds_load latency hides under the matrix pipe
//   - fp32 -> bf16 packing with single v_perm_b32 per pair
// ---------------------------------------------------------------------------

typedef __attribute__((ext_vector_type(16))) __bf16 v16bf;
typedef __attribute__((ext_vector_type(8)))  float  v8f;
typedef uint32_t v4ui __attribute__((ext_vector_type(4)));
typedef int      v8si __attribute__((ext_vector_type(8)));
typedef int      v4si __attribute__((ext_vector_type(4)));

#define B_SZ   4096
#define T_SZ   16
#define D_SZ   512
#define H_SZ   1024
#define K_SZ   (D_SZ + H_SZ)     // 1536
#define KT_N   (K_SZ / 32)       // 48 k-tiles of depth 32
#define KT_X   (D_SZ / 32)       // 16 k-tiles covering the x contribution
#define CHUNK_KT 4               // k-tiles per TDM chunk
#define NCHUNK  (KT_N / CHUNK_KT)        // 12
#define CHUNK_DW (CHUNK_KT * 8 * 256)    // 8 frags * 256 dw * 4 kt = 8192 dwords = 32KB

#if defined(__has_builtin)
#if __has_builtin(__builtin_amdgcn_tensor_load_to_lds)
#define USE_TDM 1
#endif
#endif
#ifndef USE_TDM
#define USE_TDM 0
#endif

union AFrag {
  v16bf v;
  uint32_t d[8];
  uint4 q[2];
};

__device__ __forceinline__ uint16_t f2bf_u(float f) {
  union { float f; uint32_t u; } c; c.f = f;
  uint32_t r = c.u + 0x7FFFu + ((c.u >> 16) & 1u);   // round-to-nearest-even
  return (uint16_t)(r >> 16);
}

// two fp32 -> packed bf16 pair, one v_perm_b32 (truncating)
__device__ __forceinline__ uint32_t pk2bf(float lo, float hi) {
  return __builtin_amdgcn_perm(__float_as_uint(hi), __float_as_uint(lo), 0x07060302u);
}

__device__ __forceinline__ float sigmoid_f(float x) {
  return 1.0f / (1.0f + __expf(-x));
}
__device__ __forceinline__ float tanh_f(float x) {
  return 1.0f - 2.0f / (1.0f + __expf(2.0f * x));
}

// ---------------------------------------------------------------------------
// Weight swizzle: fp32 weights -> bf16 WMMA-B fragments, grouped so that the
// 8 fragments (4 gates x 2 n-subtiles) a workgroup needs for one k-tile are
// one contiguous 8KB blob:  dword offset = ((nb*48 + kt)*8 + (g*2+sn))*256.
// Fragment-internal layout (16-bit B 32x16): lane L -> column (L%16),
//   lanes 0-15: K offsets 0..7 (v0..3), 16..23 (v4..7)
//   lanes 16-31: K offsets 8..15, 24..31; lane's 32 bytes contiguous.
// ---------------------------------------------------------------------------
struct GatePtrs {
  const float* w[4];   // wf, wi, wo, wc  [512, 1024]
  const float* u[4];   // uf, ui, uo, uc  [1024, 1024]
};

__global__ __launch_bounds__(256) void swizzle_weights(GatePtrs p, uint32_t* __restrict__ dst) {
  uint32_t tid  = blockIdx.x * 256u + threadIdx.x;   // 3,145,728 threads exactly
  uint32_t slot = tid & 255u;                        // dword within fragment
  uint32_t frag = tid >> 8;
  uint32_t f8   = frag & 7u;                         // g*2+sn
  uint32_t kt   = (frag >> 3) % KT_N;
  uint32_t nb   = (frag >> 3) / KT_N;
  uint32_t g    = f8 >> 1;
  uint32_t sn   = f8 & 1u;
  uint32_t lane = slot >> 3;
  uint32_t v    = slot & 7u;
  uint32_t hf   = lane >> 4;
  uint32_t n    = (nb * 2u + sn) * 16u + (lane & 15u);
  uint32_t kofs = (v < 4u) ? (8u * hf + 2u * v) : (16u + 8u * hf + 2u * (v - 4u));
  uint32_t k    = kt * 32u + kofs;

  float f0, f1;
  if (k < D_SZ) {
    const float* w = p.w[g];
    f0 = w[(size_t)k * H_SZ + n];
    f1 = w[(size_t)(k + 1) * H_SZ + n];
  } else {
    const float* u = p.u[g];
    uint32_t kk = k - D_SZ;
    f0 = u[(size_t)kk * H_SZ + n];
    f1 = u[(size_t)(kk + 1) * H_SZ + n];
  }
  dst[tid] = (uint32_t)f2bf_u(f0) | ((uint32_t)f2bf_u(f1) << 16);
}

// ---------------------------------------------------------------------------
// A-fragment loads (16-bit A 16x32 layout); base includes row and 8*hf.
// ---------------------------------------------------------------------------
__device__ __forceinline__ void load_a_f32(AFrag& a, const float* __restrict__ p) {
  float4 c0 = *(const float4*)(p);
  float4 c1 = *(const float4*)(p + 4);
  float4 c2 = *(const float4*)(p + 16);
  float4 c3 = *(const float4*)(p + 20);
  a.d[0] = pk2bf(c0.x, c0.y);  a.d[1] = pk2bf(c0.z, c0.w);
  a.d[2] = pk2bf(c1.x, c1.y);  a.d[3] = pk2bf(c1.z, c1.w);
  a.d[4] = pk2bf(c2.x, c2.y);  a.d[5] = pk2bf(c2.z, c2.w);
  a.d[6] = pk2bf(c3.x, c3.y);  a.d[7] = pk2bf(c3.z, c3.w);
}

__device__ __forceinline__ void load_a_bf16(AFrag& a, const __bf16* __restrict__ p) {
  a.q[0] = *(const uint4*)(p);        // 8 bf16 = 16 B
  a.q[1] = *(const uint4*)(p + 16);   // next 8 bf16 at +32 B
}

// ---------------------------------------------------------------------------
// Tensor Data Mover: 1D 32KB tile, global -> LDS.  D# per CDNA5 ISA 8.3/8.4:
// group0: count=1, lds_addr, global_addr, type=2; group1: data_size=8B,
// tensor_dim0=tile_dim0=4096 (8B units), tensor_dim1=1.
// ---------------------------------------------------------------------------
#if USE_TDM
__device__ __forceinline__ void tdm_load_chunk(uint32_t lds_addr, uint64_t gaddr) {
  v4ui g0;
  g0[0] = 1u;                                   // count=1, no gather
  g0[1] = lds_addr;                             // lds_addr[31:0]
  g0[2] = (uint32_t)(gaddr & 0xFFFFFFFFu);      // global_addr[31:0]
  g0[3] = (uint32_t)((gaddr >> 32) & 0x01FFFFFFu) | (2u << 30);  // ga[56:32] | type=2
  v8si g1;
  g1[0] = (int)(3u << 16);                      // workgroup_mask=0, data_size=3 (8B)
  g1[1] = (int)((4096u & 0xFFFFu) << 16);       // tensor_dim0[15:0] in [31:16]
  g1[2] = (int)((4096u >> 16) | (1u << 16));    // tensor_dim0[31:16] | tensor_dim1=1
  g1[3] = (int)(4096u << 16);                   // tile_dim0 = 4096 (8B units) = 32KB
  g1[4] = 0;                                    // tile_dim1 = tile_dim2 = 0 (unused)
  g1[5] = (int)4096u;                           // tensor_dim0_stride low
  g1[6] = 0;
  g1[7] = 0;
  v4si z = {0, 0, 0, 0};
#if defined(__clang_major__) && (__clang_major__ >= 23)
  v8si z8 = {0, 0, 0, 0, 0, 0, 0, 0};
  __builtin_amdgcn_tensor_load_to_lds(g0, g1, z, z, z8, 0);
#else
  __builtin_amdgcn_tensor_load_to_lds(g0, g1, z, z, 0);
#endif
}
#endif

// ---------------------------------------------------------------------------
// One LSTM timestep.  Grid: (H/32, B/256), block 256 = 8 waves (wave32).
// Wave w owns rows blockIdx.y*256 + w*32 .. +31, cols blockIdx.x*32 .. +31,
// all 4 gates -> 2(m) x 2(n) x 4(g) accumulator tiles (128 acc VGPRs/lane).
// ---------------------------------------------------------------------------
__global__ __launch_bounds__(256)
void lstm_step(const float* __restrict__ x,
               const uint32_t* __restrict__ wfrag,
               const __bf16* __restrict__ h_read,
               uint16_t* __restrict__ h_write,
               float* __restrict__ c,
               const float* __restrict__ bfp, const float* __restrict__ bip,
               const float* __restrict__ bop, const float* __restrict__ bcp,
               float* __restrict__ out,
               int t, int first, int last)
{
  __shared__ uint32_t ldsB[2][CHUNK_DW];   // 2 x 32KB double buffer

  const int lane  = threadIdx.x & 31;
  const int wave  = threadIdx.x >> 5;
  const int hf    = lane >> 4;
  const int lr    = lane & 15;
  const int rowb  = blockIdx.y * 256 + wave * 32;
  const int ncol0 = blockIdx.x * 32;

  // Bias-initialized accumulators (bias depends only on column -> splat rows)
  const float* bias[4] = { bfp, bip, bop, bcp };
  v8f acc[2][2][4];
#pragma unroll
  for (int g = 0; g < 4; ++g) {
#pragma unroll
    for (int sn = 0; sn < 2; ++sn) {
      float bv = bias[g][ncol0 + sn * 16 + lr];
#pragma unroll
      for (int mt = 0; mt < 2; ++mt)
#pragma unroll
        for (int r = 0; r < 8; ++r) acc[mt][sn][g][r] = bv;
    }
  }

  const float*  xr0 = x + ((size_t)(rowb + lr)      * T_SZ + t) * D_SZ + 8 * hf;
  const float*  xr1 = x + ((size_t)(rowb + 16 + lr) * T_SZ + t) * D_SZ + 8 * hf;
  const __bf16* hr0 = h_read + (size_t)(rowb + lr)      * H_SZ + 8 * hf;
  const __bf16* hr1 = h_read + (size_t)(rowb + 16 + lr) * H_SZ + 8 * hf;

  const int nchunk = first ? (KT_X / CHUNK_KT) : NCHUNK;   // t==0: x-only (h==0)
  const uint32_t* gchunk0 = wfrag + (size_t)blockIdx.x * KT_N * 8 * 256;

#if USE_TDM
  if (wave == 0)
    tdm_load_chunk((uint32_t)(uintptr_t)&ldsB[0][0], (uint64_t)(uintptr_t)gchunk0);
#endif

  for (int ck = 0; ck < nchunk; ++ck) {
    const uint32_t* lbuf = &ldsB[ck & 1][0];
#if USE_TDM
    if (wave == 0) __builtin_amdgcn_s_wait_tensorcnt(0);  // chunk ck landed
    __syncthreads();   // data visible; everyone done reading buf[(ck-1)&1]
    if (wave == 0 && ck + 1 < nchunk)
      tdm_load_chunk((uint32_t)(uintptr_t)&ldsB[(ck + 1) & 1][0],
                     (uint64_t)(uintptr_t)(gchunk0 + (size_t)(ck + 1) * CHUNK_DW));
#else
    __syncthreads();
    {   // cooperative fallback copy: 32KB by 256 threads
      const uint4* src = (const uint4*)(gchunk0 + (size_t)ck * CHUNK_DW);
      uint4* dstq = (uint4*)&ldsB[ck & 1][0];
#pragma unroll
      for (int i = 0; i < 8; ++i) dstq[threadIdx.x + 256 * i] = src[threadIdx.x + 256 * i];
    }
    __syncthreads();
#endif

#pragma unroll
    for (int kt4 = 0; kt4 < CHUNK_KT; ++kt4) {
      const int ktg = ck * CHUNK_KT + kt4;

      // ---- A operands for this k-tile ----
      AFrag a0, a1;
      if (ktg < KT_X) {                       // x contribution, fp32 -> bf16
        load_a_f32(a0, xr0 + ktg * 32);
        load_a_f32(a1, xr1 + ktg * 32);
        __builtin_prefetch(xr0 + ktg * 32 + 128, 0, 0);   // global_prefetch
      } else {                                // h contribution (bf16)
        const int kh0 = ktg * 32 - D_SZ;
        load_a_bf16(a0, hr0 + kh0);
        load_a_bf16(a1, hr1 + kh0);
      }

      // ---- all 8 B fragments into distinct VGPRs (16 ds_load_b128),
      //      then the 16 WMMAs: partial dscnt waits, LDS latency hidden ----
      const uint32_t* lk = lbuf + (kt4 * 8) * 256 + lane * 8;
      AFrag bfr[8];
#pragma unroll
      for (int f = 0; f < 8; ++f) {
        const uint32_t* lp = lk + f * 256;
        bfr[f].q[0] = *(const uint4*)(lp);
        bfr[f].q[1] = *(const uint4*)(lp + 4);
      }
#pragma unroll
      for (int f = 0; f < 8; ++f) {
        const int g  = f >> 1;
        const int sn = f & 1;
        acc[0][sn][g] = __builtin_amdgcn_wmma_f32_16x16x32_bf16(
            false, a0.v, false, bfr[f].v, (short)0, acc[0][sn][g], false, false);
        acc[1][sn][g] = __builtin_amdgcn_wmma_f32_16x16x32_bf16(
            false, a1.v, false, bfr[f].v, (short)0, acc[1][sn][g], false, false);
      }
    }
  }

  // ---- Cell update: one lane holds f,i,o,ch for the same (row, col) ----
#pragma unroll
  for (int mt = 0; mt < 2; ++mt) {
#pragma unroll
    for (int sn = 0; sn < 2; ++sn) {
      const int col = ncol0 + sn * 16 + lr;
#pragma unroll
      for (int r = 0; r < 8; ++r) {
        const int row = rowb + mt * 16 + 8 * hf + r;   // C/D layout: M = r + 8*(lane/16)
        const size_t idx = (size_t)row * H_SZ + col;
        float fg = sigmoid_f(acc[mt][sn][0][r]);
        float ig = sigmoid_f(acc[mt][sn][1][r]);
        float og = sigmoid_f(acc[mt][sn][2][r]);
        float cg = tanh_f(acc[mt][sn][3][r]);
        float cold = first ? 0.0f : c[idx];
        float cn = fg * cold + ig * cg;
        c[idx] = cn;
        float hn = og * tanh_f(cn);
        h_write[idx] = f2bf_u(hn);
        if (last) out[idx] = hn;
      }
    }
  }
}

// ---------------------------------------------------------------------------
// Host-side orchestration: swizzle weights once, then 16 sequential steps
// (kernel launches provide the inter-step global barrier).
// ---------------------------------------------------------------------------
extern "C" void kernel_launch(void* const* d_in, const int* in_sizes, int n_in,
                              void* d_out, int out_size, void* d_ws, size_t ws_size,
                              hipStream_t stream) {
  const float* x = (const float*)d_in[0];
  GatePtrs gp;
  gp.w[0] = (const float*)d_in[1];  gp.w[1] = (const float*)d_in[2];
  gp.w[2] = (const float*)d_in[3];  gp.w[3] = (const float*)d_in[4];
  gp.u[0] = (const float*)d_in[5];  gp.u[1] = (const float*)d_in[6];
  gp.u[2] = (const float*)d_in[7];  gp.u[3] = (const float*)d_in[8];
  const float* bfp = (const float*)d_in[9];
  const float* bip = (const float*)d_in[10];
  const float* bop = (const float*)d_in[11];
  const float* bcp = (const float*)d_in[12];

  // Workspace layout (44 MB total):
  //   [0,          12,582,912)  bf16 weight fragments (gate-fused, chunked)
  //   [12,582,912, 29,360,128)  c state  (fp32 [4096,1024])
  //   [29,360,128, 37,748,736)  h buffer 0 (bf16 [4096,1024])
  //   [37,748,736, 46,137,344)  h buffer 1
  uint8_t* ws = (uint8_t*)d_ws;
  uint32_t* wfrag = (uint32_t*)ws;
  float*    cbuf  = (float*)(ws + 12582912u);
  uint16_t* hbuf0 = (uint16_t*)(ws + 29360128u);
  uint16_t* hbuf1 = (uint16_t*)(ws + 37748736u);

  // 32 nb * 48 kt * 8 frags * 256 dwords = 3,145,728 threads
  swizzle_weights<<<dim3(12288), dim3(256), 0, stream>>>(gp, wfrag);

  dim3 grid(H_SZ / 32, B_SZ / 256);   // (32, 16)
  dim3 block(256);
  for (int t = 0; t < T_SZ; ++t) {
    uint16_t* h_in  = (t & 1) ? hbuf1 : hbuf0;
    uint16_t* h_out = (t & 1) ? hbuf0 : hbuf1;
    lstm_step<<<grid, block, 0, stream>>>(
        x, wfrag, (const __bf16*)h_in, h_out, cbuf,
        bfp, bip, bop, bcp, (float*)d_out,
        t, (t == 0) ? 1 : 0, (t == T_SZ - 1) ? 1 : 0);
  }
}
